// SAGE_87256555586107
// MI455X (gfx1250) — matrix operand order repaired and checked by reading, not compiled, not run on hardware
//
#include <hip/hip_runtime.h>
#include <hip/hip_bf16.h>

// ---------------------------------------------------------------------------
// GraphSAGE 2-layer forward for MI455X (gfx1250, wave32, WMMA)
//   layer0: mean-agg(x over E0) @ Wl0 + bl0 + x[:N1] @ Wr0, relu -> h
//   layer1: mean-agg(h over E1) @ Wl1 + bl1 + h[:N2] @ Wr1 -> log_softmax
// GEMMs: V_WMMA_F32_16X16X32_BF16, bf16 operands pre-converted off the
// matrix-pipe critical path; f32 accumulate.
// ---------------------------------------------------------------------------

#define N0_   1000000
#define N1_   50000
#define N2_   4096
#define E0_   1250000
#define E1_   40960
#define D_IN_  128
#define D_HID_ 256
#define D_OUT_ 64

typedef __attribute__((ext_vector_type(16))) __bf16 v16bf;
typedef __attribute__((ext_vector_type(8)))  __bf16 v8bf;
typedef __attribute__((ext_vector_type(8)))  float  v8f;

// f32 -> bf16 bits (round-to-nearest-even), as low 16 of a u32
__device__ __forceinline__ unsigned bfr(float f) {
  unsigned u = __builtin_bit_cast(unsigned, f);
  return (u + 0x7FFFu + ((u >> 16) & 1u)) >> 16;
}

__device__ __forceinline__ __bf16 f2bf(float f) {
  unsigned short h = (unsigned short)bfr(f);
  return __builtin_bit_cast(__bf16, h);
}

// ---------------------------------------------------------------------------
__global__ void zero_f32(float* __restrict__ p, long n) {
  long i = (long)blockIdx.x * blockDim.x + threadIdx.x;
  long stride = (long)gridDim.x * blockDim.x;
  for (; i < n; i += stride) p[i] = 0.0f;
}

// Transpose+convert weights: in [K x N] f32 row-major -> out [N x K] bf16
__global__ void wt_prep(const float* __restrict__ w, __bf16* __restrict__ wt,
                        int K, int N) {
  int i = blockIdx.x * blockDim.x + threadIdx.x;
  if (i >= K * N) return;
  int k = i / N, n = i % N;
  wt[(long)n * K + k] = f2bf(w[(long)k * N + n]);
}

// Row-wise f32 -> bf16 conversion with optional per-row mean scale (1/cnt).
// One thread = 8 consecutive elements; packed u32 stores.
__global__ void cvt_bf16_rows(const float* __restrict__ in,
                              const float* __restrict__ cnt,
                              __bf16* __restrict__ out, int rows, int d) {
  int cpr = d >> 3;
  long t = (long)blockIdx.x * blockDim.x + threadIdx.x;
  long total = (long)rows * cpr;
  if (t >= total) return;
  long row = t / cpr;
  int c = (int)(t % cpr) << 3;
  float s = cnt ? 1.0f / fmaxf(cnt[row], 1.0f) : 1.0f;
  const float4* p = reinterpret_cast<const float4*>(in + row * d + c);
  float4 a = p[0], b = p[1];
  uint4 o;
  o.x = (bfr(a.y * s) << 16) | bfr(a.x * s);
  o.y = (bfr(a.w * s) << 16) | bfr(a.z * s);
  o.z = (bfr(b.y * s) << 16) | bfr(b.x * s);
  o.w = (bfr(b.w * s) << 16) | bfr(b.z * s);
  *reinterpret_cast<uint4*>(out + row * d + c) = o;
}

// Edge scatter: one thread = one 4-float chunk of one edge; hardware f32 atomics.
__global__ void scatter_mean_accum(const float* __restrict__ x,
                                   const int* __restrict__ src,
                                   const int* __restrict__ dst,
                                   float* __restrict__ agg,
                                   float* __restrict__ cnt,
                                   int nedges, int d) {
  int chunks = d >> 2;
  long t = (long)blockIdx.x * blockDim.x + threadIdx.x;
  long total = (long)nedges * chunks;
  if (t >= total) return;
  int e = (int)(t / chunks);
  int c = (int)(t % chunks);
  int s = src[e], dd = dst[e];
  float4 v = reinterpret_cast<const float4*>(x + (long)s * d)[c];
  float* o = agg + (long)dd * d + (long)c * 4;
  unsafeAtomicAdd(o + 0, v.x);   // global_atomic_add_f32
  unsafeAtomicAdd(o + 1, v.y);
  unsafeAtomicAdd(o + 2, v.z);
  unsafeAtomicAdd(o + 3, v.w);
  if (c == 0) unsafeAtomicAdd(cnt + dd, 1.0f);
}

// A-fragment (16x32 bf16) per-lane load from a bf16 row.
// Lane layout: k already includes (lane>>4)*16; chunks at k..k+7 and k+8..k+15.
__device__ __forceinline__ v16bf load_a_frag(const __bf16* __restrict__ row, int k) {
  v8bf lo = *reinterpret_cast<const v8bf*>(row + k);        // global_load_b128
  v8bf hi = *reinterpret_cast<const v8bf*>(row + k + 8);    // global_load_b128
  return __builtin_shufflevector(lo, hi, 0, 1, 2, 3, 4, 5, 6, 7,
                                 8, 9, 10, 11, 12, 13, 14, 15);
}

// Fused SAGE linear: out = aggb @ Wl' + bias + xb @ Wr'  (one wave = 16x16 tile)
// aggb/xb: [M x K] bf16 (agg pre-scaled by 1/cnt). wlt/wrt: [Ncols x K] bf16.
template <int K, int RELU>
__global__ void sage_gemm_t(const __bf16* __restrict__ aggb,
                            const __bf16* __restrict__ xb,
                            const __bf16* __restrict__ wlt,
                            const __bf16* __restrict__ wrt,
                            const float*  __restrict__ bias,
                            float* __restrict__ out,
                            int M, int Ncols) {
  int wave   = blockIdx.x * (blockDim.x >> 5) + (threadIdx.x >> 5);
  int ntiles = Ncols >> 4;
  int mtile  = wave / ntiles;
  int ntile  = wave % ntiles;
  if (mtile * 16 >= M) return;               // wave-uniform; EXEC stays full
  int lane = threadIdx.x & 31;
  int r    = lane & 15;
  int half = lane >> 4;
  const __bf16* aRow = aggb + (long)(mtile * 16 + r) * K;
  const __bf16* xRow = xb   + (long)(mtile * 16 + r) * K;
  int col = ntile * 16 + r;
  const __bf16* wl = wlt + (long)col * K;
  const __bf16* wr = wrt + (long)col * K;
  int kbA = half * 16;                       // A lane k-offset within 32-chunk
  int kbB = half * 16;                       // B lane k-offset within 32-chunk
  __builtin_prefetch(xRow, 0, 1);            // global_prefetch_b8
  v8f acc = {0.f, 0.f, 0.f, 0.f, 0.f, 0.f, 0.f, 0.f};
#pragma unroll
  for (int k0 = 0; k0 < K; k0 += 32) {
    v16bf a1 = load_a_frag(aRow, k0 + kbA);
    v16bf b1 = *reinterpret_cast<const v16bf*>(wl + k0 + kbB);
    acc = __builtin_amdgcn_wmma_f32_16x16x32_bf16(false, a1, false, b1,
                                                  (short)0, acc, false, false);
    v16bf a2 = load_a_frag(xRow, k0 + kbA);
    v16bf b2 = *reinterpret_cast<const v16bf*>(wr + k0 + kbB);
    acc = __builtin_amdgcn_wmma_f32_16x16x32_bf16(false, a2, false, b2,
                                                  (short)0, acc, false, false);
  }
  float b = bias[col];
  // C/D layout: lanes 0-15 -> M = vgpr, lanes 16-31 -> M = vgpr + 8, N = lane%16
#pragma unroll
  for (int v = 0; v < 8; ++v) {
    float val = acc[v] + b;
    if (RELU) val = fmaxf(val, 0.0f);
    out[(long)(mtile * 16 + v + half * 8) * Ncols + col] = val;
  }
}

// log_softmax over rows of 64: one wave per row, 2 elements per lane
__global__ void logsoftmax64(const float* __restrict__ lin,
                             float* __restrict__ out, int rows) {
  int wave = blockIdx.x * (blockDim.x >> 5) + (threadIdx.x >> 5);
  if (wave >= rows) return;
  int lane = threadIdx.x & 31;
  const float* r = lin + (long)wave * 64;
  float v0 = r[lane], v1 = r[lane + 32];
  float m = fmaxf(v0, v1);
  for (int off = 16; off > 0; off >>= 1) m = fmaxf(m, __shfl_xor(m, off, 32));
  float s = expf(v0 - m) + expf(v1 - m);
  for (int off = 16; off > 0; off >>= 1) s += __shfl_xor(s, off, 32);
  float lse = m + logf(s);
  out[(long)wave * 64 + lane]      = v0 - lse;
  out[(long)wave * 64 + lane + 32] = v1 - lse;
}

// ---------------------------------------------------------------------------
extern "C" void kernel_launch(void* const* d_in, const int* in_sizes, int n_in,
                              void* d_out, int out_size, void* d_ws, size_t ws_size,
                              hipStream_t stream) {
  (void)in_sizes; (void)n_in; (void)out_size; (void)ws_size;
  const float* x   = (const float*)d_in[0];
  const int* es0   = (const int*)d_in[1];
  const int* ed0   = (const int*)d_in[2];
  const int* es1   = (const int*)d_in[3];
  const int* ed1   = (const int*)d_in[4];
  const float* Wl0 = (const float*)d_in[5];
  const float* bl0 = (const float*)d_in[6];
  const float* Wr0 = (const float*)d_in[7];
  const float* Wl1 = (const float*)d_in[8];
  const float* bl1 = (const float*)d_in[9];
  const float* Wr1 = (const float*)d_in[10];

  char* ws = (char*)d_ws;
  size_t off = 0;
  auto take = [&](size_t bytes) -> char* {
    char* p = ws + off;
    off += (bytes + 255) & ~(size_t)255;
    return p;
  };
  float*  agg0  = (float*)take((size_t)N1_ * D_IN_  * 4);
  float*  cnt0  = (float*)take((size_t)N1_ * 4);
  float*  h     = (float*)take((size_t)N1_ * D_HID_ * 4);
  float*  agg1  = (float*)take((size_t)N2_ * D_HID_ * 4);
  float*  cnt1  = (float*)take((size_t)N2_ * 4);
  float*  lin   = (float*)take((size_t)N2_ * D_OUT_ * 4);
  __bf16* wl0t  = (__bf16*)take((size_t)D_IN_  * D_HID_ * 2);
  __bf16* wr0t  = (__bf16*)take((size_t)D_IN_  * D_HID_ * 2);
  __bf16* wl1t  = (__bf16*)take((size_t)D_HID_ * D_OUT_ * 2);
  __bf16* wr1t  = (__bf16*)take((size_t)D_HID_ * D_OUT_ * 2);
  __bf16* aggb0 = (__bf16*)take((size_t)N1_ * D_IN_  * 2);
  __bf16* xb0   = (__bf16*)take((size_t)N1_ * D_IN_  * 2);
  __bf16* aggb1 = (__bf16*)take((size_t)N2_ * D_HID_ * 2);
  __bf16* hb    = (__bf16*)take((size_t)N2_ * D_HID_ * 2);

  // zero accumulators (every call; atomics accumulate)
  zero_f32<<<2048, 256, 0, stream>>>(agg0, (long)N1_ * D_IN_);
  zero_f32<<<64,   256, 0, stream>>>(cnt0, (long)N1_);
  zero_f32<<<1024, 256, 0, stream>>>(agg1, (long)N2_ * D_HID_);
  zero_f32<<<16,   256, 0, stream>>>(cnt1, (long)N2_);

  // weights -> bf16, K-contiguous
  wt_prep<<<(D_IN_ * D_HID_ + 255) / 256, 256, 0, stream>>>(Wl0, wl0t, D_IN_,  D_HID_);
  wt_prep<<<(D_IN_ * D_HID_ + 255) / 256, 256, 0, stream>>>(Wr0, wr0t, D_IN_,  D_HID_);
  wt_prep<<<(D_HID_ * D_OUT_ + 255) / 256, 256, 0, stream>>>(Wl1, wl1t, D_HID_, D_OUT_);
  wt_prep<<<(D_HID_ * D_OUT_ + 255) / 256, 256, 0, stream>>>(Wr1, wr1t, D_HID_, D_OUT_);

  // target features x[:N1] -> bf16 (independent of scatter)
  cvt_bf16_rows<<<(N1_ * (D_IN_ / 8) + 255) / 256, 256, 0, stream>>>(
      x, nullptr, xb0, N1_, D_IN_);

  // layer 0: scatter, mean+cvt, fused WMMA GEMM (relu)
  {
    long total = (long)E0_ * (D_IN_ / 4);
    scatter_mean_accum<<<(int)((total + 255) / 256), 256, 0, stream>>>(
        x, es0, ed0, agg0, cnt0, E0_, D_IN_);
  }
  cvt_bf16_rows<<<(N1_ * (D_IN_ / 8) + 255) / 256, 256, 0, stream>>>(
      agg0, cnt0, aggb0, N1_, D_IN_);
  {
    int waves = (N1_ / 16) * (D_HID_ / 16);   // 50000 tiles, exact
    sage_gemm_t<D_IN_, 1><<<waves / 8, 256, 0, stream>>>(
        aggb0, xb0, wl0t, wr0t, bl0, h, N1_, D_HID_);
  }

  // layer 1: scatter, mean+cvt, fused WMMA GEMM (no relu)
  {
    long total = (long)E1_ * (D_HID_ / 4);
    scatter_mean_accum<<<(int)((total + 255) / 256), 256, 0, stream>>>(
        h, es1, ed1, agg1, cnt1, E1_, D_HID_);
  }
  cvt_bf16_rows<<<(N2_ * (D_HID_ / 8) + 255) / 256, 256, 0, stream>>>(
      h, nullptr, hb, N2_, D_HID_);
  cvt_bf16_rows<<<(N2_ * (D_HID_ / 8) + 255) / 256, 256, 0, stream>>>(
      agg1, cnt1, aggb1, N2_, D_HID_);
  {
    int waves = (N2_ / 16) * (D_OUT_ / 16);   // 1024 tiles, exact
    sage_gemm_t<D_HID_, 0><<<waves / 8, 256, 0, stream>>>(
        aggb1, hb, wl1t, wr1t, bl1, lin, N2_, D_OUT_);
  }

  // log_softmax -> d_out
  logsoftmax64<<<(N2_ + 7) / 8, 256, 0, stream>>>(lin, (float*)d_out, N2_);
}
